// VerticalTokenMixupLayer_4595615007202
// MI455X (gfx1250) — matrix operand order repaired and checked by reading, not compiled, not run on hardware
//
#include <hip/hip_runtime.h>

typedef __attribute__((ext_vector_type(16))) __bf16 v16bf;
typedef __attribute__((ext_vector_type(8)))  float  v8f;

#define B_   64
#define N_   197
#define C_   768
#define H_   12
#define HD_  64
#define L_   4
#define KAP  16
#define NKV  261
#define NKVP 288   // padded kv length (multiple of 32)
#define NQP  208   // padded query rows (multiple of 16)
#define SCALE 0.125f

union U16 { unsigned int u[8]; v16bf v; };

__device__ __forceinline__ unsigned short f2bf(float f) {
  unsigned int u = __float_as_uint(f);
  u += 0x7fffu + ((u >> 16) & 1u);
  return (unsigned short)(u >> 16);
}

// A-operand (16x32 bf16) K-pair base for VGPR p; half = lane>>4
__device__ __forceinline__ int kbaseA(int p, int half) {
  return ((p & 3) << 1) + ((p & 4) ? 16 : 0) + (half << 3);
}
// B-operand (32x16 bf16) K-pair base for VGPR p (same k order as A)
__device__ __forceinline__ int kbaseB(int p, int half) {
  return ((p & 3) << 1) + ((p & 4) ? 16 : 0) + (half << 3);
}

// ---------------------------------------------------------------------------
// 1) Build padded bf16 kv buffer + bf16 weights
// ---------------------------------------------------------------------------
__global__ void prep_kernel(const float* __restrict__ src,
                            const float* __restrict__ vtm,
                            const float* __restrict__ Wq, const float* __restrict__ Wk,
                            const float* __restrict__ Wv, const float* __restrict__ Wp,
                            unsigned short* __restrict__ kvb,
                            unsigned short* __restrict__ wqb, unsigned short* __restrict__ wkb,
                            unsigned short* __restrict__ wvb, unsigned short* __restrict__ wpb) {
  const size_t KVE = (size_t)B_ * NKVP * C_;
  const size_t WE  = (size_t)C_ * C_;
  size_t idx = (size_t)blockIdx.x * blockDim.x + threadIdx.x;
  if (idx < KVE) {
    int c = (int)(idx % C_);
    int r = (int)((idx / C_) % NKVP);
    int b = (int)(idx / ((size_t)C_ * NKVP));
    float v = 0.f;
    if (r < N_) {
      v = src[((size_t)b * N_ + r) * C_ + c];
    } else if (r < NKV) {
      int rr = r - N_;
      int l = rr >> 4, t = rr & 15;
      v = vtm[(((size_t)l * B_ + b) * KAP + t) * C_ + c];
    }
    kvb[idx] = f2bf(v);
  } else {
    size_t e = idx - KVE;
    if (e < 4 * WE) {
      int w = (int)(e / WE);
      size_t i = e % WE;
      const float* W = (w == 0) ? Wq : (w == 1) ? Wk : (w == 2) ? Wv : Wp;
      unsigned short* O = (w == 0) ? wqb : (w == 1) ? wkb : (w == 2) ? wvb : wpb;
      O[i] = f2bf(W[i]);
    }
  }
}

// ---------------------------------------------------------------------------
// 2) Register-blocked bf16 GEMM: each wave computes a 16x64 tile (4 WMMA
//    accumulators reusing one A fragment). out[b][row][col] = A[b][row][:] . W[col][:]
//    transposed=1 stores vT[(b*C + col)*NKVP + row] (for attn @ V B-operand).
//    grid: (3, tilesM, B); block 128 (4 waves -> 16x256 per workgroup)
// ---------------------------------------------------------------------------
__global__ void gemm_bf16_kernel(const unsigned short* __restrict__ A,  // [B][NKVP][C]
                                 const unsigned short* __restrict__ W,  // [C][C] bf16 (n,k)
                                 unsigned short* __restrict__ out,
                                 int rows_out, int rows_pad, int transposed) {
  int b = blockIdx.z;
  int m0 = blockIdx.y << 4;
  int wave = threadIdx.x >> 5, lane = threadIdx.x & 31;
  int half = lane >> 4, mn = lane & 15;
  int n0 = (blockIdx.x << 8) + (wave << 6);      // 64 cols per wave
  const unsigned* Arow = (const unsigned*)A + (((size_t)b * NKVP + m0 + mn) * C_ >> 1);
  const unsigned* Wr[4];
#pragma unroll
  for (int j = 0; j < 4; ++j)
    Wr[j] = (const unsigned*)W + ((size_t)(n0 + (j << 4) + mn) * C_ >> 1);
  v8f acc[4] = {{}, {}, {}, {}};
  for (int k0 = 0; k0 < C_; k0 += 32) {
    U16 ua;
#pragma unroll
    for (int p = 0; p < 8; ++p)
      ua.u[p] = Arow[(k0 + kbaseA(p, half)) >> 1];
#pragma unroll
    for (int j = 0; j < 4; ++j) {
      U16 ub;
#pragma unroll
      for (int p = 0; p < 8; ++p)
        ub.u[p] = Wr[j][(k0 + kbaseB(p, half)) >> 1];
      acc[j] = __builtin_amdgcn_wmma_f32_16x16x32_bf16(false, ua.v, false, ub.v,
                                                       (short)0, acc[j], false, false);
    }
  }
#pragma unroll
  for (int j = 0; j < 4; ++j) {
    int col = n0 + (j << 4) + mn;
#pragma unroll
    for (int r = 0; r < 8; ++r) {
      int rr = m0 + r + (half << 3);   // row within batch
      unsigned short bv = (rr < rows_out) ? f2bf(acc[j][r]) : (unsigned short)0;
      if (!transposed) out[((size_t)b * rows_pad + rr) * C_ + col] = bv;
      else             out[((size_t)b * C_ + col) * NKVP + rr] = bv;
    }
  }
}

// ---------------------------------------------------------------------------
// 3) Fused attention: per (b, h, 16-query tile)
//    scores = q k^T * scale -> softmax (LDS) -> attn_out + P@V -> xb
// ---------------------------------------------------------------------------
__global__ void attn_kernel(const unsigned short* __restrict__ qb,
                            const unsigned short* __restrict__ kb,
                            const unsigned short* __restrict__ vT,
                            float* __restrict__ attn_out,
                            float* __restrict__ xb) {
  __shared__ __align__(16) float sc[16][NKVP];
  __shared__ __align__(16) unsigned short pbuf[16][NKVP];
  __shared__ float red[16][8];
  __shared__ float rowstat[16];
  const float NEG_INF = -__builtin_inff();

  int b = blockIdx.z, h = blockIdx.y, q0 = blockIdx.x << 4;
  int wave = threadIdx.x >> 5, lane = threadIdx.x & 31;
  int half = lane >> 4, mn = lane & 15;

  // --- scores = q @ k^T (q fragment hoisted: invariant across key tiles) ---
  {
    const unsigned* Aq = (const unsigned*)qb +
        ((((size_t)b * NQP + q0 + mn) * C_ + h * HD_) >> 1);
    U16 ua[2];
#pragma unroll
    for (int d = 0; d < 2; ++d)
#pragma unroll
      for (int p = 0; p < 8; ++p)
        ua[d].u[p] = Aq[((d << 5) + kbaseA(p, half)) >> 1];

    for (int nt = wave; nt < NKVP / 16; nt += 4) {
      const unsigned* Bk = (const unsigned*)kb +
          ((((size_t)b * NKVP + (nt << 4) + mn) * C_ + h * HD_) >> 1);
      v8f acc = {};
#pragma unroll
      for (int d = 0; d < 2; ++d) {
        U16 ub;
#pragma unroll
        for (int p = 0; p < 8; ++p)
          ub.u[p] = Bk[((d << 5) + kbaseB(p, half)) >> 1];
        acc = __builtin_amdgcn_wmma_f32_16x16x32_bf16(false, ua[d].v, false, ub.v,
                                                      (short)0, acc, false, false);
      }
#pragma unroll
      for (int r = 0; r < 8; ++r)
        sc[r + (half << 3)][(nt << 4) + mn] = acc[r] * SCALE;
    }
  }
  __syncthreads();

  // --- mask padded keys ---
  for (int i = threadIdx.x; i < 16 * (NKVP - NKV); i += blockDim.x) {
    int q = i / (NKVP - NKV), k = NKV + i % (NKVP - NKV);
    sc[q][k] = NEG_INF;
  }
  __syncthreads();

  // --- row softmax: 8 threads per row ---
  int qr = threadIdx.x >> 3, t = threadIdx.x & 7;
  const int CH = NKVP / 8;
  float mx = NEG_INF;
  for (int k = t * CH; k < t * CH + CH; ++k) mx = fmaxf(mx, sc[qr][k]);
  red[qr][t] = mx;
  __syncthreads();
  if (t == 0) {
    float m = red[qr][0];
    for (int i = 1; i < 8; ++i) m = fmaxf(m, red[qr][i]);
    rowstat[qr] = m;
  }
  __syncthreads();
  float m = rowstat[qr], s = 0.f;
  for (int k = t * CH; k < t * CH + CH; ++k) {
    float e = __expf(sc[qr][k] - m);
    sc[qr][k] = e;
    s += e;
  }
  red[qr][t] = s;
  __syncthreads();
  if (t == 0) {
    float sum = 0.f;
    for (int i = 0; i < 8; ++i) sum += red[qr][i];
    rowstat[qr] = sum;
  }
  __syncthreads();
  float inv = 1.0f / rowstat[qr];
  int qrow = q0 + qr;
  for (int k = t * CH; k < t * CH + CH; ++k) {
    float pv = sc[qr][k] * inv;
    pbuf[qr][k] = f2bf(pv);
    if (k < N_ && qrow < N_)
      attn_out[(((size_t)b * H_ + h) * N_ + qrow) * (size_t)N_ + k] = pv;
  }
  __syncthreads();

  // --- x = P @ V (wave handles one 16-wide d tile) ---
  {
    int n0 = wave << 4;
    const unsigned* Bv = (const unsigned*)vT +
        (((size_t)b * C_ + h * HD_ + n0 + mn) * NKVP >> 1);
    const unsigned* Ap = (const unsigned*)&pbuf[mn][0];
    v8f acc = {};
    for (int k0 = 0; k0 < NKVP; k0 += 32) {
      U16 ua, ub;
#pragma unroll
      for (int p = 0; p < 8; ++p) {
        ua.u[p] = Ap[(k0 + kbaseA(p, half)) >> 1];
        ub.u[p] = Bv[(k0 + kbaseB(p, half)) >> 1];
      }
      acc = __builtin_amdgcn_wmma_f32_16x16x32_bf16(false, ua.v, false, ub.v,
                                                    (short)0, acc, false, false);
    }
#pragma unroll
    for (int r = 0; r < 8; ++r) {
      int rr = q0 + r + (half << 3);
      xb[((size_t)b * NQP + rr) * C_ + h * HD_ + n0 + mn] = acc[r];
    }
  }
}

// ---------------------------------------------------------------------------
// 4) Output projection (register-blocked like gemm): out = x @ Wp^T + bp
//    grid: (3, 13, B); block 128
// ---------------------------------------------------------------------------
__global__ void proj_kernel(const float* __restrict__ xb,
                            const unsigned short* __restrict__ wpb,
                            const float* __restrict__ bp,
                            float* __restrict__ out_x) {
  int b = blockIdx.z;
  int m0 = blockIdx.y << 4;
  int wave = threadIdx.x >> 5, lane = threadIdx.x & 31;
  int half = lane >> 4, mn = lane & 15;
  int n0 = (blockIdx.x << 8) + (wave << 6);
  const float* Ax = xb + ((size_t)b * NQP + m0 + mn) * C_;
  const unsigned* Wr[4];
#pragma unroll
  for (int j = 0; j < 4; ++j)
    Wr[j] = (const unsigned*)wpb + ((size_t)(n0 + (j << 4) + mn) * C_ >> 1);
  v8f acc[4] = {{}, {}, {}, {}};
  for (int k0 = 0; k0 < C_; k0 += 32) {
    U16 ua;
#pragma unroll
    for (int p = 0; p < 8; ++p) {
      int ka = k0 + kbaseA(p, half);
      ua.u[p] = (unsigned)f2bf(Ax[ka]) | ((unsigned)f2bf(Ax[ka + 1]) << 16);
    }
#pragma unroll
    for (int j = 0; j < 4; ++j) {
      U16 ub;
#pragma unroll
      for (int p = 0; p < 8; ++p)
        ub.u[p] = Wr[j][(k0 + kbaseB(p, half)) >> 1];
      acc[j] = __builtin_amdgcn_wmma_f32_16x16x32_bf16(false, ua.v, false, ub.v,
                                                       (short)0, acc[j], false, false);
    }
  }
#pragma unroll
  for (int j = 0; j < 4; ++j) {
    int col = n0 + (j << 4) + mn;
    float bias = bp[col];
#pragma unroll
    for (int r = 0; r < 8; ++r) {
      int rr = m0 + r + (half << 3);
      if (rr < N_)
        out_x[((size_t)b * N_ + rr) * C_ + col] = acc[j][r] + bias;
    }
  }
}

// ---------------------------------------------------------------------------
// 5) Saliency mean over heads, iterative top-16 (tie -> lower index), gather
// ---------------------------------------------------------------------------
__global__ void saliency_kernel(const float* __restrict__ attn_out,
                                const float* __restrict__ src,
                                float* __restrict__ sal) {
  __shared__ float s[N_ - 1];
  __shared__ float rv[256];
  __shared__ int   ri[256];
  __shared__ int   topi[KAP];
  const float NEG_INF = -__builtin_inff();
  int b = blockIdx.x, tid = threadIdx.x;
  if (tid < N_ - 1) {
    float acc = 0.f;
    for (int h = 0; h < H_; ++h)
      acc += attn_out[(((size_t)b * H_ + h) * N_ + 0) * (size_t)N_ + (tid + 1)];
    s[tid] = acc * (1.0f / H_);
  }
  __syncthreads();
  for (int j = 0; j < KAP; ++j) {
    float v = (tid < N_ - 1) ? s[tid] : NEG_INF;
    rv[tid] = v; ri[tid] = tid;
    __syncthreads();
    for (int st = 128; st > 0; st >>= 1) {
      if (tid < st) {
        float v2 = rv[tid + st]; int i2 = ri[tid + st];
        if (v2 > rv[tid] || (v2 == rv[tid] && i2 < ri[tid])) { rv[tid] = v2; ri[tid] = i2; }
      }
      __syncthreads();
    }
    if (tid == 0) { topi[j] = ri[0]; s[ri[0]] = NEG_INF; }
    __syncthreads();
  }
  for (int e = tid; e < KAP * C_; e += 256) {
    int j = e / C_, c = e % C_;
    sal[((size_t)b * KAP + j) * C_ + c] = src[((size_t)b * N_ + topi[j]) * C_ + c];
  }
}

// ---------------------------------------------------------------------------
extern "C" void kernel_launch(void* const* d_in, const int* in_sizes, int n_in,
                              void* d_out, int out_size, void* d_ws, size_t ws_size,
                              hipStream_t stream) {
  (void)in_sizes; (void)n_in; (void)out_size; (void)ws_size;
  const float* src = (const float*)d_in[0];
  const float* vtm = (const float*)d_in[1];
  const float* Wq  = (const float*)d_in[2];
  const float* Wk  = (const float*)d_in[3];
  const float* Wv  = (const float*)d_in[4];
  const float* Wp  = (const float*)d_in[5];
  const float* bp  = (const float*)d_in[6];

  char* ws = (char*)d_ws;
  size_t o = 0;
  unsigned short* kvb = (unsigned short*)(ws + o); o += (size_t)B_ * NKVP * C_ * 2;
  unsigned short* wqb = (unsigned short*)(ws + o); o += (size_t)C_ * C_ * 2;
  unsigned short* wkb = (unsigned short*)(ws + o); o += (size_t)C_ * C_ * 2;
  unsigned short* wvb = (unsigned short*)(ws + o); o += (size_t)C_ * C_ * 2;
  unsigned short* wpb = (unsigned short*)(ws + o); o += (size_t)C_ * C_ * 2;
  unsigned short* qb  = (unsigned short*)(ws + o); o += (size_t)B_ * NQP * C_ * 2;
  unsigned short* kb  = (unsigned short*)(ws + o); o += (size_t)B_ * NKVP * C_ * 2;
  unsigned short* vT  = (unsigned short*)(ws + o); o += (size_t)B_ * C_ * NKVP * 2;
  float*          xb  = (float*)(ws + o);          o += (size_t)B_ * NQP * C_ * 4;

  float* out_x   = (float*)d_out;
  float* out_att = out_x + (size_t)B_ * N_ * C_;
  float* out_sal = out_att + (size_t)B_ * H_ * N_ * N_;

  {
    size_t total = (size_t)B_ * NKVP * C_ + 4 * (size_t)C_ * C_;
    int blocks = (int)((total + 255) / 256);
    prep_kernel<<<blocks, 256, 0, stream>>>(src, vtm, Wq, Wk, Wv, Wp,
                                            kvb, wqb, wkb, wvb, wpb);
  }
  gemm_bf16_kernel<<<dim3(3, NQP / 16, B_), 128, 0, stream>>>(kvb, wqb, qb, N_, NQP, 0);
  gemm_bf16_kernel<<<dim3(3, NKVP / 16, B_), 128, 0, stream>>>(kvb, wkb, kb, NKV, NKVP, 0);
  gemm_bf16_kernel<<<dim3(3, NKVP / 16, B_), 128, 0, stream>>>(kvb, wvb, vT, NKV, NKVP, 1);
  attn_kernel<<<dim3(NQP / 16, H_, B_), 128, 0, stream>>>(qb, kb, vT, out_att, xb);
  proj_kernel<<<dim3(3, NQP / 16, B_), 128, 0, stream>>>(xb, wpb, bp, out_x);
  saliency_kernel<<<B_, 256, 0, stream>>>(out_att, src, out_sal);
}
